// QuantSwinTransformerBlock_6442450944083
// MI455X (gfx1250) — compile-verified, matrix operand order checked by reading, not run on hardware
//
#include <hip/hip_runtime.h>
#include <hip/hip_bf16.h>
#include <math.h>

// ---------------------------------------------------------------------------
// Swin transformer block, fully fused for gfx1250 (MI455X, wave32, WMMA, TDM).
//   Kernel 0: f32 -> f16 weight conversion into d_ws (one-time, 384 KB).
//   Kernel 1: per-window (2048 WGs x 256 thr):
//             wave0 kicks a TENSOR_LOAD_TO_LDS of proj_w (overlapped DMA,
//             TDM pad feature gives a conflict-free 272B LDS pitch), then
//             LN1 + shifted gather -> qkv (WMMA) -> QK^T (WMMA) + bias + mask
//             -> softmax -> PV (WMMA) -> s_wait_tensorcnt -> proj (WMMA from
//             LDS weights) + residual scatter.  128 KB LDS/WG -> 2 WG/WGP.
//   Kernel 2: per-64-token tile: LN2 -> fc1+GELU (WMMA) -> fc2 (WMMA)
//             + residual.  Hidden (64x512) lives in LDS (~82 KB/WG).
// All LDS tiles use non-power-of-two pitches to spread the 64 LDS banks.
// ---------------------------------------------------------------------------

typedef _Float16 half_t;
typedef __attribute__((ext_vector_type(16))) _Float16 v16h;
typedef __attribute__((ext_vector_type(8)))  float    v8f;
typedef __attribute__((ext_vector_type(4)))  unsigned int uint32x4;
typedef __attribute__((ext_vector_type(8)))  int      int32x8;
typedef __attribute__((ext_vector_type(4)))  int      int32x4;

#define B_      8
#define IMGH    128
#define IMGW    128
#define C_      128
#define HEADS_  4
#define DH_     32
#define WS_     8
#define N_      64      // tokens per window
#define SHIFT_  4
#define HIDDEN_ 512
#define NWIN_X  16
#define NWIN_   256
#define QK_SCALE 0.17677669529663687f  // 1/sqrt(32)

// padded LDS pitches (avoid 64-bank conflicts on fragment gathers)
#define XW  136   // half pitch for 128-wide tiles (272 B/row)
#define SW  68    // float pitch for 64-wide logits
#define PW  72    // half pitch for 64-wide probabilities
#define GW  520   // half pitch for 512-wide hidden

__device__ __forceinline__ v8f wmma_f16(v16h a, v16h b, v8f c) {
  return __builtin_amdgcn_wmma_f32_16x16x32_f16(false, a, false, b, (short)0, c,
                                                false, false);
}

// A/B fragment loader, "row-major K-contiguous": element (r,k) at p0[r*ld+k].
// ISA 7.12.2 16-bit layout: lanes 0-15 -> K {0..7,16..23}, lanes 16-31 ->
// K {8..15,24..31}.  Also the B fragment when B[k][n] = W[n][k].
__device__ __forceinline__ v16h ldfrag_rk(const half_t* p0, int ld, int lane) {
  const int r  = lane & 15;
  const int kb = (lane >> 4) << 3;
  const half_t* p = p0 + r * ld + kb;
  v16h f;
#pragma unroll
  for (int i = 0; i < 4; ++i) {
    f[2 * i]     = p[2 * i];
    f[2 * i + 1] = p[2 * i + 1];
    f[8 + 2 * i] = p[16 + 2 * i];
    f[9 + 2 * i] = p[16 + 2 * i + 1];
  }
  return f;
}

// B fragment loader for K-major storage: element (k,n) at p0[k*ld + n].
__device__ __forceinline__ v16h ldfrag_kn(const half_t* p0, int ld, int lane) {
  const int n  = lane & 15;
  const int kb = (lane >> 4) << 3;
  v16h f;
#pragma unroll
  for (int i = 0; i < 4; ++i) {
    f[2 * i]     = p0[(kb + 2 * i) * ld + n];
    f[2 * i + 1] = p0[(kb + 2 * i + 1) * ld + n];
    f[8 + 2 * i] = p0[(16 + kb + 2 * i) * ld + n];
    f[9 + 2 * i] = p0[(16 + kb + 2 * i + 1) * ld + n];
  }
  return f;
}

__device__ __forceinline__ int region3(int r) {
  return (r < IMGH - WS_) ? 0 : ((r < IMGH - SHIFT_) ? 1 : 2);
}

// ---------------------------------------------------------------------------
// Kernel 0: weight f32 -> f16 into d_ws.
//   [0,49152) qkv_w | [49152,65536) proj_w | [65536,131072) fc1_w
//   [131072,196608) fc2_w
// ---------------------------------------------------------------------------
__global__ void convert_weights_kernel(const float* __restrict__ qkvw,
                                       const float* __restrict__ projw,
                                       const float* __restrict__ fc1w,
                                       const float* __restrict__ fc2w,
                                       half_t* __restrict__ w16) {
  int i = blockIdx.x * blockDim.x + threadIdx.x;
  if (i < 49152)        w16[i] = (half_t)qkvw[i];
  else if (i < 65536)   w16[i] = (half_t)projw[i - 49152];
  else if (i < 131072)  w16[i] = (half_t)fc1w[i - 65536];
  else if (i < 196608)  w16[i] = (half_t)fc2w[i - 131072];
}

// ---------------------------------------------------------------------------
// Kernel 1: fused windowed attention (one WG = one shifted 8x8 window).
// ---------------------------------------------------------------------------
__global__ __launch_bounds__(256) void swin_attn_kernel(
    const float* __restrict__ x,       // (8, 16384, 128)
    const float* __restrict__ rpb,     // (225, 4)
    const float* __restrict__ n1g, const float* __restrict__ n1b,
    const half_t* __restrict__ qkvw,   // (384,128) f16
    const float* __restrict__ qkvb,
    const half_t* __restrict__ projw,  // (128,128) f16 (TDM source)
    const float* __restrict__ projb,
    float* __restrict__ xprime)        // (8, 16384, 128)
{
  __shared__ __align__(16) half_t xo_s[N_ * XW];      // LN(x), later attn out
  __shared__ __align__(16) half_t q_s[N_ * XW];
  __shared__ __align__(16) half_t k_s[N_ * XW];
  __shared__ __align__(16) half_t v_s[N_ * XW];
  __shared__ __align__(16) float  s_s[N_ * SW];       // per-head logits
  __shared__ __align__(16) half_t p_s[N_ * PW];       // per-head probs
  __shared__ __align__(16) half_t projw_s[C_ * XW];   // TDM destination

  const int gw   = blockIdx.x;
  const int b    = gw >> 8;
  const int win  = gw & (NWIN_ - 1);
  const int wy   = win >> 4;
  const int wx   = win & (NWIN_X - 1);
  const int tid  = threadIdx.x;
  const int lane = tid & 31;
  const int wave = tid >> 5;

  // ---- Kick the Tensor Data Mover: proj_w (32 KB) -> LDS, overlapped ------
  // D# (ISA ch.8): 1 row of 4096 x 8-byte units; pad_enable with
  // pad_interval=5 (pad every 64 DWORDs = one 256B weight row) and
  // pad_amount=3 (+4 DWORDs) => LDS pitch 272 B = XW halves, bank-friendly.
  if (wave == 0) {
    const unsigned lds_base = (unsigned)(uintptr_t)(void*)projw_s;
    const unsigned long long ga = (unsigned long long)(uintptr_t)projw;
    uint32x4 g0;
    g0[0] = 1u;                                   // count=1, user descriptor
    g0[1] = lds_base;                             // lds_addr
    g0[2] = (unsigned)(ga & 0xffffffffu);         // global_addr[31:0]
    g0[3] = (unsigned)((ga >> 32) & 0x01ffffffu)  // global_addr[56:32]
          | (2u << 30);                           // type=2 ("image")
    int32x8 g1;
    g1[0] = (3 << 16)        // data_size = 8 bytes
          | (1 << 20)        // pad_enable
          | (5 << 22)        // pad_interval: every 64 DWORDs
          | (3 << 25);       // pad_amount: 4 DWORDs
    g1[1] = 4096 << 16;      // tensor_dim0[15:0] = 4096 (bits 79:64 window)
    g1[2] = 1 << 16;         // tensor_dim0 hi=0 | tensor_dim1 = 1
    g1[3] = 4096 << 16;      // tensor_dim1 hi=0 | tile_dim0 = 4096
    g1[4] = 1;               // tile_dim1 = 1, tile_dim2 = 0
    g1[5] = 4096;            // tensor_dim0_stride lo
    g1[6] = 4096 << 16;      // dim0_stride hi=0 | tensor_dim1_stride lo16
    g1[7] = 0;               // tensor_dim1_stride hi
    int32x4 gz  = {0, 0, 0, 0};                   // groups 2/3 disabled
    int32x8 gz8 = {0, 0, 0, 0, 0, 0, 0, 0};       // extra group (clang-23 form)
    __builtin_amdgcn_tensor_load_to_lds(g0, g1, gz, gz, gz8, 0);
  }

  // ---- Phase 1: LN1 + shifted gather (4 lanes per token, float4 loads) ----
  {
    const int tok = tid >> 2;                     // 0..63
    const int qd  = tid & 3;                      // quarter: 32 channels
    const int yi = tok >> 3, xi = tok & 7;
    const int R  = (wy * WS_ + yi + SHIFT_) & (IMGH - 1);
    const int Cc = (wx * WS_ + xi + SHIFT_) & (IMGW - 1);
    const float4* src =
        (const float4*)(x + ((size_t)b * (IMGH * IMGW) + R * IMGW + Cc) * C_)
        + qd * 8;
    float4 vb[8];
    float s = 0.f;
#pragma unroll
    for (int i = 0; i < 8; ++i) {
      vb[i] = src[i];
      s += vb[i].x + vb[i].y + vb[i].z + vb[i].w;
    }
    s += __shfl_xor(s, 1);
    s += __shfl_xor(s, 2);
    const float m = s * (1.f / C_);
    float vv = 0.f;
#pragma unroll
    for (int i = 0; i < 8; ++i) {
      float a0 = vb[i].x - m, a1 = vb[i].y - m, a2 = vb[i].z - m, a3 = vb[i].w - m;
      vv += a0 * a0 + a1 * a1 + a2 * a2 + a3 * a3;
    }
    vv += __shfl_xor(vv, 1);
    vv += __shfl_xor(vv, 2);
    const float inv = rsqrtf(vv * (1.f / C_) + 1e-5f);
    const int c0 = qd * 32;
#pragma unroll
    for (int i = 0; i < 8; ++i) {
      const int c = c0 + i * 4;
      xo_s[tok * XW + c + 0] = (half_t)((vb[i].x - m) * inv * n1g[c + 0] + n1b[c + 0]);
      xo_s[tok * XW + c + 1] = (half_t)((vb[i].y - m) * inv * n1g[c + 1] + n1b[c + 1]);
      xo_s[tok * XW + c + 2] = (half_t)((vb[i].z - m) * inv * n1g[c + 2] + n1b[c + 2]);
      xo_s[tok * XW + c + 3] = (half_t)((vb[i].w - m) * inv * n1g[c + 3] + n1b[c + 3]);
    }
  }
  __syncthreads();

  // ---- Phase 2: qkv = LN(x) @ qkv_w^T + qkv_b  (64x384, WMMA) -------------
  for (int t = wave; t < 96; t += 8) {            // 4 M-tiles x 24 N-tiles
    const int mt = t / 24, nt = t % 24;
    __builtin_prefetch(qkvw + (nt * 16) * C_, 0, 0);
    v8f acc = {};
#pragma unroll
    for (int kt = 0; kt < 4; ++kt) {
      v16h a  = ldfrag_rk(xo_s + mt * 16 * XW + kt * 32, XW, lane);
      v16h bf = ldfrag_rk(qkvw + nt * 16 * C_ + kt * 32, C_, lane);
      acc = wmma_f16(a, bf, acc);
    }
    const int n  = lane & 15;
    const int rb = (lane >> 4) << 3;
    const int col = nt * 16 + n;
    const float bias = qkvb[col];
    half_t* dst; int c2; float scl = 1.f;
    if (col < C_)          { dst = q_s; c2 = col;          scl = QK_SCALE; }
    else if (col < 2 * C_) { dst = k_s; c2 = col - C_;     }
    else                   { dst = v_s; c2 = col - 2 * C_; }
#pragma unroll
    for (int r = 0; r < 8; ++r) {
      const int row = mt * 16 + rb + r;
      dst[row * XW + c2] = (half_t)((acc[r] + bias) * scl);
    }
  }
  __syncthreads();

  // ---- Phase 3+4: per head, S = QK^T + bias + mask; softmax; O = P V ------
  for (int hh = 0; hh < HEADS_; ++hh) {
    for (int t = wave; t < 16; t += 8) {          // S tiles (K = 32)
      const int mt = t >> 2, nt = t & 3;
      v16h a  = ldfrag_rk(q_s + mt * 16 * XW + hh * DH_, XW, lane);
      v16h bf = ldfrag_rk(k_s + nt * 16 * XW + hh * DH_, XW, lane);
      v8f acc = {};
      acc = wmma_f16(a, bf, acc);
      const int n  = lane & 15;
      const int rb = (lane >> 4) << 3;
      const int mcol = nt * 16 + n;               // key token
      const int ky = mcol >> 3, kx = mcol & 7;
      const int idk = region3(wy * WS_ + ky) * 3 + region3(wx * WS_ + kx);
#pragma unroll
      for (int r = 0; r < 8; ++r) {
        const int nrow = mt * 16 + rb + r;        // query token
        const int qy = nrow >> 3, qx = nrow & 7;
        const int idx = (qy - ky + 7) * 15 + (qx - kx + 7);
        const float bias = rpb[idx * HEADS_ + hh];
        const int idq = region3(wy * WS_ + qy) * 3 + region3(wx * WS_ + qx);
        const float mval = (idq != idk) ? -100.f : 0.f;
        s_s[nrow * SW + mcol] = acc[r] + bias + mval;
      }
    }
    __syncthreads();

    // softmax: 4 lanes per query row, shfl_xor quad reductions
    {
      const int row = tid >> 2, qd = tid & 3;
      float* rp = s_s + row * SW + qd * 16;
      float mx = -3.4e38f;
#pragma unroll
      for (int j = 0; j < 16; ++j) mx = fmaxf(mx, rp[j]);
      mx = fmaxf(mx, __shfl_xor(mx, 1));
      mx = fmaxf(mx, __shfl_xor(mx, 2));
      float e[16];
      float sum = 0.f;
#pragma unroll
      for (int j = 0; j < 16; ++j) { e[j] = __expf(rp[j] - mx); sum += e[j]; }
      sum += __shfl_xor(sum, 1);
      sum += __shfl_xor(sum, 2);
      const float inv = 1.f / sum;
      half_t* pp = p_s + row * PW + qd * 16;
#pragma unroll
      for (int j = 0; j < 16; ++j) pp[j] = (half_t)(e[j] * inv);
    }
    __syncthreads();

    // O = P V  (64x32, K = 64)
    for (int t = wave; t < 8; t += 8) {
      const int mt = t >> 1, dt = t & 1;
      v8f acc = {};
#pragma unroll
      for (int kt = 0; kt < 2; ++kt) {
        v16h a  = ldfrag_rk(p_s + mt * 16 * PW + kt * 32, PW, lane);
        v16h bf = ldfrag_kn(v_s + (kt * 32) * XW + hh * DH_ + dt * 16, XW, lane);
        acc = wmma_f16(a, bf, acc);
      }
      const int n  = lane & 15;
      const int rb = (lane >> 4) << 3;
#pragma unroll
      for (int r = 0; r < 8; ++r) {
        const int row = mt * 16 + rb + r;
        xo_s[row * XW + hh * DH_ + dt * 16 + n] = (half_t)acc[r];
      }
    }
    __syncthreads();
  }

  // ---- TDM completion: proj weights are now in LDS ------------------------
  if (wave == 0) __builtin_amdgcn_s_wait_tensorcnt((short)0);
  __syncthreads();

  // ---- Phase 5: proj + residual, scatter back through reverse shift -------
  for (int t = wave; t < 32; t += 8) {            // 4 M-tiles x 8 N-tiles
    const int mt = t >> 3, nt = t & 7;
    v8f acc = {};
#pragma unroll
    for (int kt = 0; kt < 4; ++kt) {
      v16h a  = ldfrag_rk(xo_s + mt * 16 * XW + kt * 32, XW, lane);
      v16h bf = ldfrag_rk(projw_s + nt * 16 * XW + kt * 32, XW, lane);
      acc = wmma_f16(a, bf, acc);
    }
    const int n  = lane & 15;
    const int rb = (lane >> 4) << 3;
    const int ch = nt * 16 + n;
    const float pb = projb[ch];
#pragma unroll
    for (int r = 0; r < 8; ++r) {
      const int tok = mt * 16 + rb + r;
      const int yi = tok >> 3, xi = tok & 7;
      const int R  = (wy * WS_ + yi + SHIFT_) & (IMGH - 1);
      const int Cc = (wx * WS_ + xi + SHIFT_) & (IMGW - 1);
      const size_t off = ((size_t)b * (IMGH * IMGW) + R * IMGW + Cc) * C_ + ch;
      xprime[off] = x[off] + acc[r] + pb;
    }
  }
}

// ---------------------------------------------------------------------------
// Kernel 2: fused MLP (one WG = 64 tokens), in-place x' -> x' + mlp(LN2(x')).
// ---------------------------------------------------------------------------
__global__ __launch_bounds__(256) void swin_mlp_kernel(
    const float* xin,                    // aliases outp
    const float* __restrict__ n2g, const float* __restrict__ n2b,
    const half_t* __restrict__ fc1w,     // (512,128) f16
    const float* __restrict__ fc1b,
    const half_t* __restrict__ fc2w,     // (128,512) f16
    const float* __restrict__ fc2b,
    float* outp)
{
  __shared__ __align__(16) half_t h_s[N_ * XW];   // LN2 tile
  __shared__ __align__(16) half_t g_s[N_ * GW];   // gelu(fc1)

  const int tb   = blockIdx.x * N_;
  const int tid  = threadIdx.x;
  const int lane = tid & 31;
  const int wave = tid >> 5;

  // LN2 (4 lanes per token)
  {
    const int tok = tid >> 2;
    const int qd  = tid & 3;
    const float4* src = (const float4*)(xin + (size_t)(tb + tok) * C_) + qd * 8;
    float4 vb[8];
    float s = 0.f;
#pragma unroll
    for (int i = 0; i < 8; ++i) {
      vb[i] = src[i];
      s += vb[i].x + vb[i].y + vb[i].z + vb[i].w;
    }
    s += __shfl_xor(s, 1);
    s += __shfl_xor(s, 2);
    const float m = s * (1.f / C_);
    float vv = 0.f;
#pragma unroll
    for (int i = 0; i < 8; ++i) {
      float a0 = vb[i].x - m, a1 = vb[i].y - m, a2 = vb[i].z - m, a3 = vb[i].w - m;
      vv += a0 * a0 + a1 * a1 + a2 * a2 + a3 * a3;
    }
    vv += __shfl_xor(vv, 1);
    vv += __shfl_xor(vv, 2);
    const float inv = rsqrtf(vv * (1.f / C_) + 1e-5f);
    const int c0 = qd * 32;
#pragma unroll
    for (int i = 0; i < 8; ++i) {
      const int c = c0 + i * 4;
      h_s[tok * XW + c + 0] = (half_t)((vb[i].x - m) * inv * n2g[c + 0] + n2b[c + 0]);
      h_s[tok * XW + c + 1] = (half_t)((vb[i].y - m) * inv * n2g[c + 1] + n2b[c + 1]);
      h_s[tok * XW + c + 2] = (half_t)((vb[i].z - m) * inv * n2g[c + 2] + n2b[c + 2]);
      h_s[tok * XW + c + 3] = (half_t)((vb[i].w - m) * inv * n2g[c + 3] + n2b[c + 3]);
    }
  }
  __syncthreads();

  // fc1 + exact GELU  (64x512, K=128)
  for (int t = wave; t < 128; t += 8) {           // 4 M-tiles x 32 N-tiles
    const int mt = t >> 5, nt = t & 31;
    __builtin_prefetch(fc1w + (nt * 16) * C_, 0, 0);
    v8f acc = {};
#pragma unroll
    for (int kt = 0; kt < 4; ++kt) {
      v16h a  = ldfrag_rk(h_s + mt * 16 * XW + kt * 32, XW, lane);
      v16h bf = ldfrag_rk(fc1w + nt * 16 * C_ + kt * 32, C_, lane);
      acc = wmma_f16(a, bf, acc);
    }
    const int n  = lane & 15;
    const int rb = (lane >> 4) << 3;
    const int col = nt * 16 + n;
    const float bias = fc1b[col];
#pragma unroll
    for (int r = 0; r < 8; ++r) {
      const int row = mt * 16 + rb + r;
      const float v = acc[r] + bias;
      const float g = 0.5f * v * (1.f + erff(v * 0.70710678118654752f));
      g_s[row * GW + col] = (half_t)g;
    }
  }
  __syncthreads();

  // fc2 + residual  (64x128, K=512)
  for (int t = wave; t < 32; t += 8) {            // 4 M-tiles x 8 N-tiles
    const int mt = t >> 3, nt = t & 7;
    __builtin_prefetch(fc2w + (nt * 16) * HIDDEN_, 0, 0);
    v8f acc = {};
#pragma unroll
    for (int kt = 0; kt < 16; ++kt) {
      v16h a  = ldfrag_rk(g_s + mt * 16 * GW + kt * 32, GW, lane);
      v16h bf = ldfrag_rk(fc2w + nt * 16 * HIDDEN_ + kt * 32, HIDDEN_, lane);
      acc = wmma_f16(a, bf, acc);
    }
    const int n  = lane & 15;
    const int rb = (lane >> 4) << 3;
    const int col = nt * 16 + n;
    const float bias = fc2b[col];
#pragma unroll
    for (int r = 0; r < 8; ++r) {
      const int tok = tb + mt * 16 + rb + r;
      const size_t off = (size_t)tok * C_ + col;
      outp[off] = xin[off] + acc[r] + bias;       // same-thread read->write
    }
  }
}

// ---------------------------------------------------------------------------
extern "C" void kernel_launch(void* const* d_in, const int* in_sizes, int n_in,
                              void* d_out, int out_size, void* d_ws, size_t ws_size,
                              hipStream_t stream) {
  const float* x      = (const float*)d_in[0];
  const float* rpb    = (const float*)d_in[1];
  const float* n1g    = (const float*)d_in[2];
  const float* n1b    = (const float*)d_in[3];
  const float* qkvw32 = (const float*)d_in[4];
  const float* qkvb   = (const float*)d_in[5];
  const float* projw32= (const float*)d_in[6];
  const float* projb  = (const float*)d_in[7];
  const float* n2g    = (const float*)d_in[8];
  const float* n2b    = (const float*)d_in[9];
  const float* fc1w32 = (const float*)d_in[10];
  const float* fc1b   = (const float*)d_in[11];
  const float* fc2w32 = (const float*)d_in[12];
  const float* fc2b   = (const float*)d_in[13];
  float* out = (float*)d_out;

  half_t* w16     = (half_t*)d_ws;
  half_t* qkvw16  = w16;
  half_t* projw16 = w16 + 49152;
  half_t* fc1w16  = w16 + 65536;
  half_t* fc2w16  = w16 + 131072;

  convert_weights_kernel<<<768, 256, 0, stream>>>(qkvw32, projw32, fc1w32,
                                                  fc2w32, w16);
  swin_attn_kernel<<<2048, 256, 0, stream>>>(x, rpb, n1g, n1b, qkvw16, qkvb,
                                             projw16, projb, out);
  swin_mlp_kernel<<<2048, 256, 0, stream>>>(out, n2g, n2b, fc1w16, fc1b,
                                            fc2w16, fc2b, out);
}
